// GMTForCausalLM_90460601189047
// MI455X (gfx1250) — matrix operand order repaired and checked by reading, not compiled, not run on hardware
//
#include <hip/hip_runtime.h>
#include <hip/hip_bf16.h>
#include <math.h>

typedef __attribute__((ext_vector_type(16))) __bf16 v16bf;
typedef __attribute__((ext_vector_type(8)))  __bf16 v8bf;
typedef __attribute__((ext_vector_type(8)))  float  v8f;

// ---------------------------------------------------------------- utilities
__device__ __forceinline__ float wave_sum(float v) {
    #pragma unroll
    for (int m = 16; m > 0; m >>= 1) v += __shfl_xor(v, m, 32);
    return v;
}

// ------------------------------------------------- WMMA NT GEMM  C = A*W^T+b
// A:[M,K] bf16 row-major, W:[N,K] bf16 row-major. M%16==0, N%64==0, K%32==0.
// One wave computes a 16x64 tile; A fragment reused across 4 WMMAs.
// Optional f32 output C and/or bf16 mirror Cbf (for downstream WMMA GEMMs).
__global__ void __launch_bounds__(256)
gemm_bf16_nt(const __bf16* __restrict__ A, int lda,
             const __bf16* __restrict__ W, int ldw,
             const float* __restrict__ bias,
             float* __restrict__ C, __bf16* __restrict__ Cbf, int ldc,
             int M, int N, int K, int gelu)
{
    const int lane = threadIdx.x & 31;
    const int wv   = threadIdx.x >> 5;
    const int nt64 = N >> 6;
    const int total = (M >> 4) * nt64;
    int tile = blockIdx.x * 8 + wv;
    if (tile >= total) return;
    const int mt  = tile / nt64;
    const int nt4 = tile % nt64;
    const int half = lane >> 4;
    const int l15  = lane & 15;

    // A 16x32 bf16 k order chosen here (same as B): lanes 0-15: K 0..15; lanes 16-31: K 16..31
    const __bf16* arow = A + (size_t)(mt * 16 + l15) * lda + half * 16;
    // B 32x16 bf16 k order chosen here (same as A): lane = N col; lanes 0-15: K 0..15, lanes 16-31: K 16..31
    const __bf16* wrow = W + (size_t)(nt4 * 64 + l15) * ldw + half * 16;

    v8f acc[4] = {v8f{}, v8f{}, v8f{}, v8f{}};

    for (int k0 = 0; k0 < K; k0 += 32) {
        v8bf a0 = *(const v8bf*)(arow + k0);
        v8bf a1 = *(const v8bf*)(arow + k0 + 8);
        v16bf af = __builtin_shufflevector(a0, a1,
                     0,1,2,3,4,5,6,7,8,9,10,11,12,13,14,15);
        #pragma unroll
        for (int j = 0; j < 4; ++j) {
            v16bf bf = *(const v16bf*)(wrow + (size_t)j * 16 * ldw + k0);
            acc[j] = __builtin_amdgcn_wmma_f32_16x16x32_bf16(
                false, af, false, bf, (short)0, acc[j], false, false);
        }
    }

    // C/D f32 layout: VGPR r -> M = mt*16 + half*8 + r, N = ntile*16 + l15
    #pragma unroll
    for (int r = 0; r < 8; ++r) {
        size_t mrow = (size_t)(mt * 16 + half * 8 + r) * ldc;
        #pragma unroll
        for (int j = 0; j < 4; ++j) {
            int n = (nt4 * 4 + j) * 16 + l15;
            float v = acc[j][r] + bias[n];
            if (gelu) v = 0.5f * v * (1.0f + erff(v * 0.70710678118654752f));
            if (C)   C[mrow + n]   = v;
            if (Cbf) Cbf[mrow + n] = (__bf16)v;
        }
    }
}

// ------------------------------------------------------------- small kernels
__global__ void __launch_bounds__(256)
cvt_f32_bf16(const float* __restrict__ in, __bf16* __restrict__ out, int n8)
{
    int i = blockIdx.x * 256 + threadIdx.x;
    if (i >= n8) return;
    const float4* p = (const float4*)(in + (size_t)i * 8);
    float4 a = p[0], b = p[1];
    v8bf o;
    o[0]=(__bf16)a.x; o[1]=(__bf16)a.y; o[2]=(__bf16)a.z; o[3]=(__bf16)a.w;
    o[4]=(__bf16)b.x; o[5]=(__bf16)b.y; o[6]=(__bf16)b.z; o[7]=(__bf16)b.w;
    *(v8bf*)(out + (size_t)i * 8) = o;
}

__global__ void __launch_bounds__(256)
rownorm_kernel(const float* __restrict__ X, float* __restrict__ norms, int rows)
{
    int row = blockIdx.x * 8 + (threadIdx.x >> 5);
    if (row >= rows) return;
    int lane = threadIdx.x & 31;
    const float* x = X + (size_t)row * 512;
    float s = 0.f;
    #pragma unroll
    for (int i = 0; i < 16; ++i) { float t = x[lane + i * 32]; s += t * t; }
    s = wave_sum(s);
    if (lane == 0) norms[row] = fmaxf(sqrtf(s), 1e-12f);
}

__global__ void __launch_bounds__(256)
gather_rows(const float* __restrict__ table, const int* __restrict__ ids,
            float* __restrict__ out)
{
    int row = blockIdx.x;
    const float* src = table + (size_t)ids[row] * 512;
    float* dst = out + (size_t)row * 512;
    dst[threadIdx.x]       = src[threadIdx.x];
    dst[threadIdx.x + 256] = src[threadIdx.x + 256];
}

// cosine sims over N=32 neighbors, top-8 (smallest-index tiebreak), mean agg
__global__ void __launch_bounds__(256)
topk_agg_kernel(const int* __restrict__ edge_ids, const int* __restrict__ neigh_ids,
                const float* __restrict__ table, const float* __restrict__ norms,
                __bf16* __restrict__ agg)
{
    __shared__ float s_rel[512];
    __shared__ float s_sims[32];
    __shared__ int   s_nid[32];
    __shared__ int   s_sel[8];

    const int be = blockIdx.x;
    const int t = threadIdx.x, lane = t & 31, wv = t >> 5;

    const int eid = edge_ids[be];
    const float* relrow = table + (size_t)eid * 512;
    s_rel[t] = relrow[t];
    s_rel[t + 256] = relrow[t + 256];
    __syncthreads();
    const float relnorm = norms[eid];

    for (int nn = wv; nn < 32; nn += 8) {
        int nid = neigh_ids[be * 32 + nn];
        const float* nrow = table + (size_t)nid * 512;
        float d = 0.f;
        #pragma unroll
        for (int i = 0; i < 16; ++i) d += nrow[lane + i * 32] * s_rel[lane + i * 32];
        d = wave_sum(d);
        if (lane == 0) { s_sims[nn] = d / (relnorm * norms[nid]); s_nid[nn] = nid; }
    }
    __syncthreads();

    if (wv == 0) {
        float v = s_sims[lane];
        #pragma unroll
        for (int r = 0; r < 8; ++r) {
            float bv = v; int bi = lane;
            #pragma unroll
            for (int m = 16; m > 0; m >>= 1) {
                float ov = __shfl_xor(bv, m, 32);
                int   oi = __shfl_xor(bi, m, 32);
                if (ov > bv || (ov == bv && oi < bi)) { bv = ov; bi = oi; }
            }
            if (lane == 0) s_sel[r] = bi;
            if (lane == bi) v = -3.4028235e38f;
        }
    }
    __syncthreads();

    float a0 = 0.f, a1 = 0.f;
    #pragma unroll
    for (int r = 0; r < 8; ++r) {
        const float* nrow = table + (size_t)s_nid[s_sel[r]] * 512;
        a0 += nrow[t];
        a1 += nrow[t + 256];
    }
    __bf16* dst = agg + (size_t)be * 512;
    dst[t]       = (__bf16)(a0 * 0.125f);
    dst[t + 256] = (__bf16)(a1 * 0.125f);
}

// O = LN(X + Rb)*g + b (optionally * mask[row]); bf16 mirror for WMMA consumers
__global__ void __launch_bounds__(256)
add_ln_kernel(const float* __restrict__ X, const float* __restrict__ Rb,
              const float* __restrict__ g, const float* __restrict__ bb,
              const float* __restrict__ mask, float* __restrict__ O,
              __bf16* __restrict__ Obf, int rows)
{
    int row = blockIdx.x * 8 + (threadIdx.x >> 5);
    if (row >= rows) return;
    int lane = threadIdx.x & 31;
    const float* x = X + (size_t)row * 512;
    const float* r = Rb + (size_t)row * 512;
    float vals[16];
    float s = 0.f;
    #pragma unroll
    for (int i = 0; i < 16; ++i) {
        float v = x[lane + i * 32] + r[lane + i * 32];
        vals[i] = v; s += v;
    }
    float mu = wave_sum(s) * (1.0f / 512.0f);
    float q = 0.f;
    #pragma unroll
    for (int i = 0; i < 16; ++i) { float d = vals[i] - mu; q += d * d; }
    float rstd = rsqrtf(wave_sum(q) * (1.0f / 512.0f) + 1e-5f);
    float mk = mask ? mask[row] : 1.0f;
    float* o = O + (size_t)row * 512;
    __bf16* ob = Obf + (size_t)row * 512;
    #pragma unroll
    for (int i = 0; i < 16; ++i) {
        int c = lane + i * 32;
        float v = ((vals[i] - mu) * rstd * g[c] + bb[c]) * mk;
        o[c] = v;
        ob[c] = (__bf16)v;
    }
}

// memory-token cross attention: one block per (b, m); head softmax per wave
__global__ void __launch_bounds__(256)
attn_kernel(const float* __restrict__ Q, const float* __restrict__ Kb,
            const float* __restrict__ Vb, const float* __restrict__ mask,
            __bf16* __restrict__ Om)
{
    __shared__ float s_q[512];
    __shared__ float s_sc[8][256];
    __shared__ float s_den[8];

    const int b = blockIdx.x >> 5;
    const int m = blockIdx.x & 31;
    const int t = threadIdx.x;
    const int lane = t & 31, wv = t >> 5;

    s_q[t] = Q[m * 512 + t];
    s_q[t + 256] = Q[m * 512 + 256 + t];
    __syncthreads();

    const float* krow = Kb + (size_t)(b * 256 + t) * 512;
    const float mk = mask[b * 256 + t];
    #pragma unroll
    for (int h = 0; h < 8; ++h) {
        float s = 0.f;
        #pragma unroll
        for (int d = 0; d < 64; ++d) s += s_q[h * 64 + d] * krow[h * 64 + d];
        s_sc[h][t] = (mk == 0.f) ? -3.4028235e38f : s * 0.125f;
    }
    __syncthreads();

    {   // wave wv: softmax of head wv over E=256
        float mx = -3.4028235e38f;
        #pragma unroll
        for (int i = 0; i < 8; ++i) mx = fmaxf(mx, s_sc[wv][lane + i * 32]);
        #pragma unroll
        for (int mm = 16; mm > 0; mm >>= 1) mx = fmaxf(mx, __shfl_xor(mx, mm, 32));
        float sum = 0.f;
        #pragma unroll
        for (int i = 0; i < 8; ++i) {
            int e = lane + i * 32;
            float p = __expf(s_sc[wv][e] - mx);
            s_sc[wv][e] = p;
            sum += p;
        }
        sum = wave_sum(sum);
        if (lane == 0) s_den[wv] = sum;
    }
    __syncthreads();

    const int h0 = t >> 6, h1 = 4 + (t >> 6);
    float o0 = 0.f, o1 = 0.f;
    const float* vb = Vb + (size_t)b * 256 * 512;
    for (int e = 0; e < 256; ++e) {
        const float* vrow = vb + (size_t)e * 512;
        o0 += s_sc[h0][e] * vrow[t];
        o1 += s_sc[h1][e] * vrow[t + 256];
    }
    __bf16* orow = Om + (size_t)(b * 32 + m) * 512;
    orow[t]       = (__bf16)(o0 / s_den[h0]);
    orow[t + 256] = (__bf16)(o1 / s_den[h1]);
}

// zero memory-token rows for batches with no edges (after out-proj bias)
__global__ void __launch_bounds__(256)
noedge_zero_kernel(const float* __restrict__ mask, __bf16* __restrict__ memo)
{
    __shared__ float red[256];
    const int b = blockIdx.x, t = threadIdx.x;
    red[t] = mask[b * 256 + t];
    __syncthreads();
    for (int s = 128; s > 0; s >>= 1) {
        if (t < s) red[t] += red[t + s];
        __syncthreads();
    }
    if (red[0] != 0.f) return;
    for (int i = t; i < 32 * 512; i += 256) memo[(size_t)b * 32 * 512 + i] = (__bf16)0.0f;
}

// --------------------------------------------------------------------- host
static void launch_gemm(const __bf16* A, int lda, const __bf16* W, int ldw,
                        const float* bias, float* C, __bf16* Cbf, int ldc,
                        int M, int N, int K, int gelu, hipStream_t stream)
{
    int tiles = (M / 16) * (N / 64);
    int blocks = (tiles + 7) / 8;
    hipLaunchKernelGGL(gemm_bf16_nt, dim3(blocks), dim3(256), 0, stream,
                       A, lda, W, ldw, bias, C, Cbf, ldc, M, N, K, gelu);
}

static void launch_cvt(const float* in, __bf16* out, size_t n, hipStream_t stream)
{
    int n8 = (int)(n / 8);
    hipLaunchKernelGGL(cvt_f32_bf16, dim3((n8 + 255) / 256), dim3(256), 0, stream,
                       in, out, n8);
}

extern "C" void kernel_launch(void* const* d_in, const int* in_sizes, int n_in,
                              void* d_out, int out_size, void* d_ws, size_t ws_size,
                              hipStream_t stream)
{
    (void)in_sizes; (void)n_in; (void)out_size; (void)ws_size;
    const int*   edge_ids  = (const int*)d_in[0];
    const int*   neigh_ids = (const int*)d_in[1];
    const float* edge_mask = (const float*)d_in[2];
    const float* rel_emb   = (const float*)d_in[3];
    const float* rp_w      = (const float*)d_in[4];
    const float* rp_b      = (const float*)d_in[5];
    const float* ly_vw  = (const float*)d_in[6];
    const float* ly_vb  = (const float*)d_in[7];
    const float* ly_ow  = (const float*)d_in[8];
    const float* ly_ob  = (const float*)d_in[9];
    const float* ly_n1g = (const float*)d_in[10];
    const float* ly_n1b = (const float*)d_in[11];
    const float* ly_n2g = (const float*)d_in[12];
    const float* ly_n2b = (const float*)d_in[13];
    const float* ly_w1  = (const float*)d_in[14];
    const float* ly_b1  = (const float*)d_in[15];
    const float* ly_w2  = (const float*)d_in[16];
    const float* ly_b2  = (const float*)d_in[17];
    const float* mem_q  = (const float*)d_in[18];
    const float* t_qw = (const float*)d_in[19];
    const float* t_qb = (const float*)d_in[20];
    const float* t_kw = (const float*)d_in[21];
    const float* t_kb = (const float*)d_in[22];
    const float* t_vw = (const float*)d_in[23];
    const float* t_vb = (const float*)d_in[24];
    const float* t_ow = (const float*)d_in[25];
    const float* t_ob = (const float*)d_in[26];
    const float* proj_w = (const float*)d_in[27];
    const float* proj_b = (const float*)d_in[28];
    float* out = (float*)d_out;

    const int BE = 8 * 256;      // 2048 edges

    // ---- f32 workspace region
    float* ws = (float*)d_ws;
    size_t off = 0;
    float* proj_all = ws + off; off += (size_t)2048 * 512;
    float* pnorms   = ws + off; off += 2048;
    float* states   = ws + off; off += (size_t)BE * 512;
    float* tmpB     = ws + off; off += (size_t)BE * 512;
    float* kbuf     = ws + off; off += (size_t)BE * 512;
    float* vbuf     = ws + off; off += (size_t)BE * 512;
    float* qbuf     = ws + off; off += (size_t)32 * 512;

    // ---- bf16 workspace region (all offsets multiple of 16 elements)
    __bf16* bws = (__bf16*)(ws + off);
    size_t boff = 0;
    __bf16* abf      = bws + boff; boff += (size_t)2000 * 768;  // rel_emb / mem_q cvt
    __bf16* wbf      = bws + boff; boff += (size_t)4096 * 512;  // per-GEMM weight cvt
    __bf16* aggbf    = bws + boff; boff += (size_t)BE * 512;
    __bf16* statesbf = bws + boff; boff += (size_t)BE * 512;
    __bf16* tmpAbf   = bws + boff; boff += (size_t)BE * 512;
    __bf16* hbufbf   = bws + boff; boff += (size_t)BE * 2048;
    __bf16* membbf   = bws + boff; boff += (size_t)256 * 512;
    __bf16* memobf   = bws + boff; boff += (size_t)256 * 512;

    // 1. project the whole relation table once: [2000,768] @ [512,768]^T
    launch_cvt(rel_emb, abf, (size_t)2000 * 768, stream);
    launch_cvt(rp_w,    wbf, (size_t)512 * 768, stream);
    launch_gemm(abf, 768, wbf, 768, rp_b, proj_all, nullptr, 512, 2000, 512, 768, 0, stream);
    // 2. row norms for cosine similarity
    hipLaunchKernelGGL(rownorm_kernel, dim3(250), dim3(256), 0, stream, proj_all, pnorms, 2000);
    // 3. edge states = gathered projected relations
    hipLaunchKernelGGL(gather_rows, dim3(BE), dim3(256), 0, stream, proj_all, edge_ids, states);
    // 4. neighbor sims -> top-8 -> mean aggregate (bf16 for WMMA consumers)
    hipLaunchKernelGGL(topk_agg_kernel, dim3(BE), dim3(256), 0, stream,
                       edge_ids, neigh_ids, proj_all, pnorms, aggbf);
    // 5. relation-context layers
    for (int l = 0; l < 2; ++l) {
        launch_cvt(ly_vw + (size_t)l * 512 * 512, wbf, (size_t)512 * 512, stream);
        launch_gemm(aggbf, 512, wbf, 512, ly_vb + l * 512,
                    nullptr, tmpAbf, 512, BE, 512, 512, 0, stream);
        launch_cvt(ly_ow + (size_t)l * 512 * 512, wbf, (size_t)512 * 512, stream);
        launch_gemm(tmpAbf, 512, wbf, 512, ly_ob + l * 512,
                    tmpB, nullptr, 512, BE, 512, 512, 0, stream);
        hipLaunchKernelGGL(add_ln_kernel, dim3(BE / 8), dim3(256), 0, stream,
                           states, tmpB, ly_n1g + l * 512, ly_n1b + l * 512,
                           (const float*)nullptr, states, statesbf, BE);
        launch_cvt(ly_w1 + (size_t)l * 2048 * 512, wbf, (size_t)2048 * 512, stream);
        launch_gemm(statesbf, 512, wbf, 512, ly_b1 + l * 2048,
                    nullptr, hbufbf, 2048, BE, 2048, 512, 1, stream);
        launch_cvt(ly_w2 + (size_t)l * 512 * 2048, wbf, (size_t)512 * 2048, stream);
        launch_gemm(hbufbf, 2048, wbf, 2048, ly_b2 + l * 512,
                    tmpB, nullptr, 512, BE, 512, 2048, 0, stream);
        hipLaunchKernelGGL(add_ln_kernel, dim3(BE / 8), dim3(256), 0, stream,
                           states, tmpB, ly_n2g + l * 512, ly_n2b + l * 512,
                           edge_mask, states, statesbf, BE);
    }
    // 6. tokenizer q/k/v
    launch_cvt(mem_q, abf, (size_t)32 * 512, stream);
    launch_cvt(t_qw, wbf, (size_t)512 * 512, stream);
    launch_gemm(abf, 512, wbf, 512, t_qb, qbuf, nullptr, 512, 32, 512, 512, 0, stream);
    launch_cvt(t_kw, wbf, (size_t)512 * 512, stream);
    launch_gemm(statesbf, 512, wbf, 512, t_kb, kbuf, nullptr, 512, BE, 512, 512, 0, stream);
    launch_cvt(t_vw, wbf, (size_t)512 * 512, stream);
    launch_gemm(statesbf, 512, wbf, 512, t_vb, vbuf, nullptr, 512, BE, 512, 512, 0, stream);
    // 7. cross attention -> memory tokens (bf16 for out-proj GEMM)
    hipLaunchKernelGGL(attn_kernel, dim3(256), dim3(256), 0, stream,
                       qbuf, kbuf, vbuf, edge_mask, membbf);
    // 8. out-proj, no-edge zeroing, final LLM projection
    launch_cvt(t_ow, wbf, (size_t)512 * 512, stream);
    launch_gemm(membbf, 512, wbf, 512, t_ob, nullptr, memobf, 512, 256, 512, 512, 0, stream);
    hipLaunchKernelGGL(noedge_zero_kernel, dim3(8), dim3(256), 0, stream, edge_mask, memobf);
    launch_cvt(proj_w, wbf, (size_t)4096 * 512, stream);
    launch_gemm(memobf, 512, wbf, 512, proj_b, out, nullptr, 4096, 256, 4096, 512, 0, stream);
}
